// LoRALinear_29566554866428
// MI455X (gfx1250) — matrix operand order repaired and checked by reading, not compiled, not run on hardware
//
#include <hip/hip_runtime.h>
#include <stdint.h>

// ---------- types ----------
typedef __attribute__((ext_vector_type(16))) __bf16       v16bf;
typedef __attribute__((ext_vector_type(8)))  float        v8f;
typedef __attribute__((ext_vector_type(4)))  unsigned int u32x4;
typedef __attribute__((ext_vector_type(2)))  unsigned int u32x2;
typedef __attribute__((ext_vector_type(4)))  float        f32x4;
typedef __attribute__((ext_vector_type(8)))  int          i32x8;
typedef __attribute__((ext_vector_type(4)))  int          i32x4;

#define D_IN   4096
#define D_OUT  4096
#define M_TOK  4096        // 2 * 2048
#define LORA_SCALE 2.0f    // alpha/rank = 16/8

// fp32 -> bf16, round-to-nearest-even
__device__ __forceinline__ unsigned short f2bf(float f) {
  unsigned int u = __float_as_uint(f);
  u += 0x7FFFu + ((u >> 16) & 1u);
  return (unsigned short)(u >> 16);
}

// ---------------------------------------------------------------------------
// Kernel 1: convert x (fp32) -> bf16 workspace. 256 thr * 8 elem = 2048/block
// ---------------------------------------------------------------------------
__global__ __launch_bounds__(256) void cvt_x_kernel(const float* __restrict__ x,
                                                    unsigned short* __restrict__ xh) {
  const int i = blockIdx.x * 2048 + threadIdx.x * 8;
  f32x4 a = *(const f32x4*)(x + i);
  f32x4 b = *(const f32x4*)(x + i + 4);
  u32x4 o;
  o.x = (unsigned)f2bf(a.x) | ((unsigned)f2bf(a.y) << 16);
  o.y = (unsigned)f2bf(a.z) | ((unsigned)f2bf(a.w) << 16);
  o.z = (unsigned)f2bf(b.x) | ((unsigned)f2bf(b.y) << 16);
  o.w = (unsigned)f2bf(b.z) | ((unsigned)f2bf(b.w) << 16);
  *(u32x4*)(xh + i) = o;
}

// ---------------------------------------------------------------------------
// Kernel 2: Wh[n][k] = bf16(base[n][k] + 2.0 * sum_r B[n][r] * A[r][k])
// ---------------------------------------------------------------------------
__global__ __launch_bounds__(256) void fuse_w_kernel(const float* __restrict__ base,
                                                     const float* __restrict__ lA,
                                                     const float* __restrict__ lB,
                                                     unsigned short* __restrict__ wh) {
  const int n = blockIdx.x;
  float bv[8];
#pragma unroll
  for (int r = 0; r < 8; ++r) bv[r] = lB[n * 8 + r] * LORA_SCALE;

  const float* brow = base + (size_t)n * D_IN;
  unsigned short* wrow = wh + (size_t)n * D_IN;
  const int k0 = threadIdx.x * 16;
#pragma unroll
  for (int g = 0; g < 4; ++g) {
    const int k = k0 + g * 4;
    f32x4 acc = *(const f32x4*)(brow + k);
#pragma unroll
    for (int r = 0; r < 8; ++r) {
      f32x4 av = *(const f32x4*)(lA + r * D_IN + k);
      acc.x += bv[r] * av.x; acc.y += bv[r] * av.y;
      acc.z += bv[r] * av.z; acc.w += bv[r] * av.w;
    }
    u32x2 o;
    o.x = (unsigned)f2bf(acc.x) | ((unsigned)f2bf(acc.y) << 16);
    o.y = (unsigned)f2bf(acc.z) | ((unsigned)f2bf(acc.w) << 16);
    *(u32x2*)(wrow + k) = o;
  }
}

// ---------------------------------------------------------------------------
// Kernel 3: bf16 WMMA GEMM with TDM (tensor_load_to_lds) tile feeding.
// Block tile 128x128, BK=32. 8 waves: 4 (M) x 2 (N); wave tile 32x64.
// TDM pads each 64B tile row with 16B -> LDS row stride 80B (LDSS=40 halves),
// which makes the per-lane 16B fragment loads bank-conflict-free.
// ---------------------------------------------------------------------------
#define BM 128
#define BN 128
#define BK 32
#define LDSS 40   // halves per LDS row (32 + 8 pad), produced by TDM pad fields

union FragBF { v16bf v; u32x4 q[2]; };

// Issue one 2D TDM tile load: tile_dim0 = BK elems (K), tile_dim1 = rows,
// tensor row stride = D_IN elems, element size 2B, LDS pad 16B per 64B row.
__device__ __forceinline__ void tdm_load_2d(unsigned lds_off,
                                            const unsigned short* gptr) {
  const unsigned long long ga = (unsigned long long)(uintptr_t)gptr;
  u32x4 g0;
  g0.x = 1u;                                                 // count=1 (valid user D#)
  g0.y = lds_off;                                            // lds_addr (bytes)
  g0.z = (unsigned)ga;                                       // global_addr[31:0]
  g0.w = (unsigned)((ga >> 32) & 0x1FFFFFFu) | (2u << 30);   // addr[56:32] | type=2
  i32x8 g1;
  g1[0] = (1 << 16)      // data_size = 1 -> 2 bytes
        | (1 << 20)      // pad_enable
        | (3 << 22)      // pad_interval: 16 DWORDs (64B) between pads
        | (3 << 25);     // pad_amount: 4 DWORDs (16B)
  g1[1] = (int)(4096u << 16);   // tensor_dim0[15:0]=4096 in bits[79:48] low half
  g1[2] = (int)(4096u << 16);   // tensor_dim0 hi=0 | tensor_dim1[15:0]=4096
  g1[3] = (int)(32u << 16);     // tensor_dim1 hi=0 | tile_dim0=32 (BK)
  g1[4] = 128;                  // tile_dim1=128 rows, tile_dim2=0
  g1[5] = 4096;                 // tensor_dim0_stride lo32 = D_IN
  g1[6] = 0;                    // stride0 hi | stride1 lo
  g1[7] = 0;                    // stride1 hi
  const i32x4 gz4 = {0, 0, 0, 0};
  const i32x8 gz8 = {0, 0, 0, 0, 0, 0, 0, 0};
  __builtin_amdgcn_tensor_load_to_lds(g0, g1, gz4, gz4, gz8, 0);
}

__global__ __launch_bounds__(256) void lora_gemm_kernel(
    const unsigned short* __restrict__ xh,
    const unsigned short* __restrict__ wh,
    const float* __restrict__ bias,
    float* __restrict__ out) {
  __shared__ __align__(16) unsigned short ldsA[2][BM * LDSS];
  __shared__ __align__(16) unsigned short ldsB[2][BN * LDSS];

  const int tid  = threadIdx.x;
  const int lane = tid & 31;
  const int wave = tid >> 5;
  const int wm   = wave >> 1;   // 0..3 -> M offset wm*32
  const int wn   = wave & 1;    // 0..1 -> N offset wn*64
  const int half = lane >> 4;   // lane half selects K sub-chunk per ISA layout
  const int l16  = lane & 15;

  const int mblock = blockIdx.y * BM;
  const int nblock = blockIdx.x * BN;

  // LDS byte offsets of the four tile buffers (flat addr low 32 bits = LDS offset)
  const unsigned oA[2] = { (unsigned)(uintptr_t)&ldsA[0][0],
                           (unsigned)(uintptr_t)&ldsA[1][0] };
  const unsigned oB[2] = { (unsigned)(uintptr_t)&ldsB[0][0],
                           (unsigned)(uintptr_t)&ldsB[1][0] };

  const unsigned short* gA = xh + (size_t)mblock * D_IN;
  const unsigned short* gB = wh + (size_t)nblock * D_IN;

  // preload K-tile 0 into buffer 0 via TDM (wave 0 drives the DMA)
  if (wave == 0) {
    tdm_load_2d(oA[0], gA);
    tdm_load_2d(oB[0], gB);
    __builtin_amdgcn_s_wait_tensorcnt(0);
  }
  __syncthreads();
  asm volatile("" ::: "memory");   // LDS contents written by TDM are opaque

  v8f acc[2][4];
  const v8f vzero = {0.f, 0.f, 0.f, 0.f, 0.f, 0.f, 0.f, 0.f};
#pragma unroll
  for (int i = 0; i < 2; ++i)
#pragma unroll
    for (int j = 0; j < 4; ++j) acc[i][j] = vzero;

  const int KT = D_IN / BK;   // 128
  for (int kt = 0; kt < KT; ++kt) {
    const int buf = kt & 1;
    const bool has_next = (kt + 1) < KT;

    // kick off DMA for the next K-tile into the other buffer
    if (has_next && wave == 0) {
      const int nb = buf ^ 1;
      tdm_load_2d(oA[nb], gA + (kt + 1) * BK);
      tdm_load_2d(oB[nb], gB + (kt + 1) * BK);
    }

    // A fragments (16x32): lane half h holds K {8h..8h+7, 16+8h..16+8h+7}
    FragBF afrag[2];
#pragma unroll
    for (int i = 0; i < 2; ++i) {
      const unsigned short* p = &ldsA[buf][(wm * 32 + i * 16 + l16) * LDSS];
      afrag[i].q[0] = *(const u32x4*)(p + 8 * half);
      afrag[i].q[1] = *(const u32x4*)(p + 16 + 8 * half);
    }
    // B fragments (32x16): lane half h holds K {8h..8h+7, 16+8h..16+8h+7} of col l16
    FragBF bfrag[4];
#pragma unroll
    for (int j = 0; j < 4; ++j) {
      const unsigned short* p = &ldsB[buf][(wn * 64 + j * 16 + l16) * LDSS];
      bfrag[j].q[0] = *(const u32x4*)(p + 8 * half);
      bfrag[j].q[1] = *(const u32x4*)(p + 16 + 8 * half);
    }

#pragma unroll
    for (int i = 0; i < 2; ++i)
#pragma unroll
      for (int j = 0; j < 4; ++j)
        acc[i][j] = __builtin_amdgcn_wmma_f32_16x16x32_bf16(
            false, afrag[i].v, false, bfrag[j].v,
            (short)0, acc[i][j], false, false);

    // publish the next buffer: DMA must be complete before the barrier opens it
    if (has_next && wave == 0)
      __builtin_amdgcn_s_wait_tensorcnt(0);
    __syncthreads();
    asm volatile("" ::: "memory");
  }

  // epilogue: C/D layout -> lane l (<16): N=l, M=v; lane l+16: N=l, M=8+v
#pragma unroll
  for (int j = 0; j < 4; ++j) {
    const int n = nblock + wn * 64 + j * 16 + l16;
    const float bv = bias[n];
#pragma unroll
    for (int i = 0; i < 2; ++i) {
      const int m0 = mblock + wm * 32 + i * 16 + half * 8;
      float* po = out + (size_t)m0 * D_OUT + n;
#pragma unroll
      for (int v = 0; v < 8; ++v)
        po[(size_t)v * D_OUT] = acc[i][j][v] + bv;
    }
  }
}

// ---------------------------------------------------------------------------
// launch
// ---------------------------------------------------------------------------
extern "C" void kernel_launch(void* const* d_in, const int* in_sizes, int n_in,
                              void* d_out, int out_size, void* d_ws, size_t ws_size,
                              hipStream_t stream) {
  const float* x    = (const float*)d_in[0];   // [2,2048,4096]
  const float* base = (const float*)d_in[1];   // [4096,4096]
  const float* lA   = (const float*)d_in[2];   // [8,4096]
  const float* lB   = (const float*)d_in[3];   // [4096,8]
  const float* bias = (const float*)d_in[4];   // [4096]
  float* out        = (float*)d_out;           // [4096,4096]

  unsigned short* xh = (unsigned short*)d_ws;                       // 32 MB
  unsigned short* wh = xh + (size_t)M_TOK * D_IN;                   // 32 MB

  cvt_x_kernel<<<(M_TOK * D_IN) / 2048, 256, 0, stream>>>(x, xh);
  fuse_w_kernel<<<D_OUT, 256, 0, stream>>>(base, lA, lB, wh);

  dim3 grid(D_OUT / BN, M_TOK / BM);
  lora_gemm_kernel<<<grid, 256, 0, stream>>>(xh, wh, bias, out);
}
